// LearnablePixelwiseAnisoJBU3D_89721866813723
// MI455X (gfx1250) — matrix operand order repaired and checked by reading, not compiled, not run on hardware
//
#include <hip/hip_runtime.h>
#include <hip/hip_bf16.h>

// ---------------------------------------------------------------------------
// LearnablePixelwiseAnisoJBU3D for MI455X (gfx1250, wave32, WMMA, TDM)
//
// 16 hi-res voxels (2x2x4, two low-res base cells stacked in Z) share one
// 7x7x8 low-res neighborhood (K=392, padded to 416 = 13 chunks of 32).
// num(16x32) = W(16x392) x F(392x32) via v_wmma_f32_16x16x32_f16.
// Neighborhood boxes are DMA'd into LDS by the Tensor Data Mover.
// K ordering: K = pair*8 + (dz+3), pair = (dx+3)*7 + (dy+3); every 8
// consecutive K share (dx,dy), so decode/clamp hoists 8x.
//
// Workspace layout (requires ws_size >= 4 MiB):
//   [0, 1MiB)    : sig records, 32768 x 8 f32 {i2x,i2y,i2z,i2r,g0,g1,g2,seff}
//   [1MiB, 3MiB) : featT, (h,w,d, c-interleaved) 32768 x 16 u32 (half2 c,c+16)
//   [3MiB, 4MiB) : R2 map, 64^3 f32
// ---------------------------------------------------------------------------

typedef __attribute__((ext_vector_type(16))) _Float16 v16h;
typedef __attribute__((ext_vector_type(8)))  float    v8f;
typedef unsigned int u32;
typedef unsigned long long u64;
typedef __attribute__((ext_vector_type(4))) u32 u32x4;
typedef __attribute__((ext_vector_type(8))) int i32x8;
typedef __attribute__((ext_vector_type(4))) int i32x4;

#define HL 32
#define NEG_INF (-1.0e9f)

// Per-wave LDS tile layout (bytes)
#define FEAT_OFF 0
#define FEAT_BYTES (7 * 7 * 8 * 32 * 2)        // 25088
#define SIG_OFF  FEAT_BYTES
#define SIG_BYTES (7 * 7 * 8 * 8 * 4)          // 12544
#define DEN_OFF  (FEAT_BYTES + SIG_BYTES)      // 37632
#define PER_WAVE_LDS (DEN_OFF + 64)            // 37696
#define WAVES_PER_BLOCK 4
#define DYN_LDS (PER_WAVE_LDS * WAVES_PER_BLOCK)

__device__ __forceinline__ int clampi(int v, int lo, int hi) {
  return v < lo ? lo : (v > hi ? hi : v);
}

// ---------------------------------------------------------------------------
// Prep 1: reciprocal sigma terms + guide 2x downsample (= 8-avg), packed as
// 8-float records; slot 7 = sigma_eff = max(sx,sy,sz) (for the R2 map).
// ---------------------------------------------------------------------------
__global__ void prep_sig_kernel(const float* __restrict__ sxr,
                                const float* __restrict__ syr,
                                const float* __restrict__ szr,
                                const float* __restrict__ srr,
                                const float* __restrict__ guide,
                                float* __restrict__ sig) {
  int idx = blockIdx.x * blockDim.x + threadIdx.x;  // 32768
  int h = idx >> 10, w = (idx >> 5) & 31, d = idx & 31;
  float sx = fmaxf(__expf(sxr[idx]), 1e-6f);
  float sy = fmaxf(__expf(syr[idx]), 1e-6f);
  float sz = fmaxf(__expf(szr[idx]), 1e-6f);
  float sr = fmaxf(__expf(srr[idx]), 1e-6f);
  int gb = ((2 * h) * 64 + 2 * w) * 64 + 2 * d;
  float g[3];
#pragma unroll
  for (int c = 0; c < 3; c++) {
    const float* p = guide + c * 262144 + gb;
    g[c] = 0.125f * (p[0] + p[1] + p[64] + p[65] +
                     p[4096] + p[4097] + p[4160] + p[4161]);
  }
  float* r = sig + (size_t)idx * 8;
  r[0] = 0.5f / (sx * sx);
  r[1] = 0.5f / (sy * sy);
  r[2] = 0.5f / (sz * sz);
  r[3] = 1.0f / (2.0f * sr * sr + 1e-8f);
  r[4] = g[0]; r[5] = g[1]; r[6] = g[2];
  r[7] = fmaxf(sx, fmaxf(sy, sz));
}

// ---------------------------------------------------------------------------
// Prep 2: R2 map at high-res (trilinear upsample of sigma_eff).
// ---------------------------------------------------------------------------
__global__ void prep_r2_kernel(const float* __restrict__ sig,
                               float* __restrict__ r2map) {
  int idx = blockIdx.x * blockDim.x + threadIdx.x;  // 262144
  int X = idx >> 12, Y = (idx >> 6) & 63, Z = idx & 63;
  float u = (X + 0.5f) * 0.5f - 0.5f;
  float v = (Y + 0.5f) * 0.5f - 0.5f;
  float w = (Z + 0.5f) * 0.5f - 0.5f;
  int i0 = (int)floorf(u), j0 = (int)floorf(v), k0 = (int)floorf(w);
  float fu = u - i0, fv = v - j0, fw = w - k0;
  int i0c = clampi(i0, 0, 31), i1c = clampi(i0 + 1, 0, 31);
  int j0c = clampi(j0, 0, 31), j1c = clampi(j0 + 1, 0, 31);
  int k0c = clampi(k0, 0, 31), k1c = clampi(k0 + 1, 0, 31);
  auto S = [&](int a, int b, int c) {
    return sig[((size_t)((a * 32 + b) * 32 + c)) * 8 + 7];
  };
  float c00 = S(i0c, j0c, k0c) * (1 - fw) + S(i0c, j0c, k1c) * fw;
  float c01 = S(i0c, j1c, k0c) * (1 - fw) + S(i0c, j1c, k1c) * fw;
  float c10 = S(i1c, j0c, k0c) * (1 - fw) + S(i1c, j0c, k1c) * fw;
  float c11 = S(i1c, j1c, k0c) * (1 - fw) + S(i1c, j1c, k1c) * fw;
  float c0 = c00 * (1 - fv) + c01 * fv;
  float c1 = c10 * (1 - fv) + c11 * fv;
  float s = c0 * (1 - fu) + c1 * fu;
  float R = fminf(fmaxf(ceilf(2.0f * s), 1.0f), 3.0f);
  r2map[idx] = R * R;
}

// ---------------------------------------------------------------------------
// Prep 3: feat (c,h,w,d) f32 -> (h,w,d) x 16 dwords, dword j = half2(c=j,
// c=j+16). One ds_load_b32 then yields both WMMA N-tiles for a point.
// ---------------------------------------------------------------------------
__global__ void prep_feat_kernel(const float* __restrict__ feat,
                                 u32* __restrict__ featT32) {
  int idx = blockIdx.x * blockDim.x + threadIdx.x;  // 32768
  u32* dst = featT32 + (size_t)idx * 16;
#pragma unroll
  for (int j = 0; j < 16; j++) {
    _Float16 a = (_Float16)feat[(size_t)j * 32768 + idx];
    _Float16 b = (_Float16)feat[(size_t)(j + 16) * 32768 + idx];
    u32 ua = (u32)__builtin_bit_cast(unsigned short, a);
    u32 ub = (u32)__builtin_bit_cast(unsigned short, b);
    dst[j] = ua | (ub << 16);
  }
}

// ---------------------------------------------------------------------------
// TDM: [7 x 7 x tile0] box from a 3D tensor (dim0 fused innermost). D# per
// CDNA5 ISA; clang-23 6-arg builtin form.
// ---------------------------------------------------------------------------
#if __has_builtin(__builtin_amdgcn_tensor_load_to_lds)
#define HAVE_TDM 1
__device__ __forceinline__ void tdm_load_box(u32 lds_addr, u64 gaddr,
                                             u32 data_size_code,  // 1=2B, 2=4B
                                             u32 tensor_dim0, u32 tensor_dim1,
                                             u32 tensor_dim2, u32 tile_dim0,
                                             u32 dim0_stride, u32 dim1_stride,
                                             u32 dim2_stride) {
  u32x4 g0;
  g0.x = 1u;                                    // count=1, user descriptor
  g0.y = lds_addr;                              // lds_addr [63:32]
  g0.z = (u32)(gaddr & 0xffffffffull);          // global_addr lo
  g0.w = (u32)((gaddr >> 32) & 0x01ffffffull) | (2u << 30);  // addr hi | type=2
  i32x8 g1;
  g1[0] = (int)(data_size_code << 16);          // wg_mask=0, data_size
  g1[1] = (int)((tensor_dim0 & 0xffffu) << 16);
  g1[2] = (int)(((tensor_dim0 >> 16) & 0xffffu) |
                ((tensor_dim1 & 0xffffu) << 16));
  g1[3] = (int)(((tensor_dim1 >> 16) & 0xffffu) |
                ((tile_dim0 & 0xffffu) << 16));
  g1[4] = (int)(7u | (7u << 16));               // tile_dim1=7, tile_dim2=7
  g1[5] = (int)dim0_stride;
  g1[6] = (int)(((dim1_stride & 0xffffu) << 16));
  g1[7] = (int)(dim1_stride >> 16);
  i32x4 g2;
  g2[0] = (int)tensor_dim2;
  g2[1] = 0;
  g2[2] = (int)dim2_stride;
  g2[3] = (int)(dim2_stride >> 16) & 0xffff;
  i32x4 g3;
  g3[0] = 0; g3[1] = 0; g3[2] = 0; g3[3] = 0;
  i32x8 gx;
  gx[0] = 0; gx[1] = 0; gx[2] = 0; gx[3] = 0;
  gx[4] = 0; gx[5] = 0; gx[6] = 0; gx[7] = 0;
  __builtin_amdgcn_tensor_load_to_lds(g0, g1, g2, g3, gx, 0);
}
#else
#define HAVE_TDM 0
#endif

// ---------------------------------------------------------------------------
// Main kernel: one wave per 16-voxel tile; 13 x 2 WMMA f32_16x16x32_f16.
// ---------------------------------------------------------------------------
__global__ __launch_bounds__(128) void jbu_main_kernel(
    const float* __restrict__ guide, const float* __restrict__ sig,
    const u32* __restrict__ featT, const float* __restrict__ r2map,
    float* __restrict__ out) {
  extern __shared__ __align__(16) unsigned char smem[];
  const int lane = threadIdx.x & 31;
  const int wave = threadIdx.x >> 5;
  const int tile = blockIdx.x * WAVES_PER_BLOCK + wave;
  const int ui = tile >> 9;          // low-res h cell
  const int vi = (tile >> 4) & 31;   // low-res w cell
  const int wp = tile & 15;          // z cell pair -> cells {2wp, 2wp+1}

  const int h0 = clampi(ui - 3, 0, HL - 7);
  const int w0 = clampi(vi - 3, 0, HL - 7);
  const int z0 = clampi(2 * wp - 3, 0, HL - 8);
  const u32 waveBase = (u32)(wave * PER_WAVE_LDS);

#if HAVE_TDM
  {
    // feat box: dim0 = fused (d,c) = 1024 f16, w stride 1024, h stride 32768.
    u64 gf = (u64)(uintptr_t)featT +
             2ull * (((u64)(h0 * 32 + w0)) * 1024ull + (u64)z0 * 32ull);
    tdm_load_box(waveBase + FEAT_OFF, gf, 1u,
                 1024u, 32u, 32u, /*tile0=*/8u * 32u,
                 1024u, 32768u, 32768u);
    // sig box: dim0 = fused (d,8f32) = 256 f32, w stride 256, h stride 8192.
    u64 gs = (u64)(uintptr_t)sig +
             4ull * (((u64)(h0 * 32 + w0)) * 256ull + (u64)z0 * 8ull);
    tdm_load_box(waveBase + SIG_OFF, gs, 2u,
                 256u, 32u, 32u, /*tile0=*/8u * 8u,
                 256u, 8192u, 8192u);
  }
#if __has_builtin(__builtin_amdgcn_s_wait_tensorcnt)
  __builtin_amdgcn_s_wait_tensorcnt(0);
#endif
#else
  for (int i = lane; i < 7 * 7 * 8; i += 32) {
    int bi = i / 56, bj = (i / 8) % 7, bk = i & 7;
    int gi = ((h0 + bi) * 32 + (w0 + bj)) * 32 + (z0 + bk);
    float4* sd = (float4*)(smem + waveBase + SIG_OFF + i * 32);
    const float4* ss = (const float4*)(sig + (size_t)gi * 8);
    sd[0] = ss[0]; sd[1] = ss[1];
    float4* fd = (float4*)(smem + waveBase + FEAT_OFF + i * 64);
    const float4* fs = (const float4*)(featT + (size_t)gi * 16);
    fd[0] = fs[0]; fd[1] = fs[1]; fd[2] = fs[2]; fd[3] = fs[3];
  }
#endif

  // Per-lane voxel (A row): m = lane % 16.
  const int m = lane & 15;
  const int x2 = m & 1, y2 = (m >> 1) & 1, z4 = m >> 2;
  const int X = 2 * ui + x2, Y = 2 * vi + y2, Z = 4 * wp + z4;
  const int zhalf = z4 >> 1;
  const float Xf = (float)X, Yf = (float)Y, Zf = (float)Z;
  const int hrIdx = X * 4096 + Y * 64 + Z;
  const float R2m = r2map[hrIdx];
  const float gh0 = guide[hrIdx];
  const float gh1 = guide[262144 + hrIdx];
  const float gh2 = guide[524288 + hrIdx];

  // Evaluate the 8 log-weights of one (dx,dy) pair-group (dz = -3..+4),
  // branchless. p in [0,52); pairs >= 49 are padding (forced NEG_INF).
  auto eval_pairgroup = [&](int p, auto&& perEl) {
    int pvalid = (p < 49) ? 1 : 0;
    u32 ps = (u32)(pvalid ? p : 0);
    int dx = (int)(ps / 7u) - 3;
    int dy = (int)(ps % 7u) - 3;
    int Ui = clampi(ui + dx, 0, HL - 1);
    int Vi = clampi(vi + dy, 0, HL - 1);
    int rowBase = ((Ui - h0) * 7 + (Vi - w0)) * 8;
    float dxh = Xf - (2.0f * Ui + 0.5f);
    float dyh = Yf - (2.0f * Vi + 0.5f);
    float dxh2 = dxh * dxh, dyh2 = dyh * dyh;
    int dxy2 = dx * dx + dy * dy;
#pragma unroll
    for (int e = 0; e < 8; e++) {
      int dzp = e - 3;
      int Wi = clampi(2 * wp + dzp, 0, HL - 1);
      const float4* rec = (const float4*)(smem + waveBase + SIG_OFF +
                                          ((rowBase + (Wi - z0)) << 5));
      float4 s0 = rec[0];  // 1/2sx2, 1/2sy2, 1/2sz2, 1/(2sr2+eps)
      float4 s1 = rec[1];  // g0, g1, g2, seff
      int dze = dzp - zhalf;
      int r2 = dxy2 + dze * dze;
      int oki = pvalid & (int)((u32)(dze + 3) <= 6u) &
                (((float)r2 <= R2m) ? 1 : 0);
      float dzh = Zf - (2.0f * Wi + 0.5f);
      float lws = dxh2 * s0.x + dyh2 * s0.y + dzh * dzh * s0.z;
      float d0 = gh0 - s1.x, d1 = gh1 - s1.y, d2 = gh2 - s1.z;
      float lw = -(lws + (d0 * d0 + d1 * d1 + d2 * d2) * s0.w);
      lw = oki ? lw : NEG_INF;
      perEl(e, lw);
    }
  };

  // A-fragment pair bases (16-bit A 16x32 layout): lanes 0-15 hold K
  // {0..7,16..23} -> pairs {4q, 4q+2}; lanes 16-31 hold K {8..15,24..31}
  // -> pairs {4q+1, 4q+3}.
  const int hA = (lane < 16) ? 0 : 1;
  const int bb = (lane < 16) ? 0 : 1;  // B pair base (K {0..7,16..23} / {8..15,24..31})
  const int n = lane & 15;             // B/C column (channel)

  // Pass 1: per-voxel max of log-weights.
  float mmax = NEG_INF;
  for (int q = 0; q < 13; q++) {
    eval_pairgroup(4 * q + hA,
                   [&](int e, float lw) { mmax = fmaxf(mmax, lw); });
    eval_pairgroup(4 * q + 2 + hA,
                   [&](int e, float lw) { mmax = fmaxf(mmax, lw); });
  }
  mmax = fmaxf(mmax, __shfl_xor(mmax, 16, 32));

  // Pass 2: softmax weights -> A; LDS feature dwords -> B0/B1; WMMA.
  v8f acc0 = {};
  v8f acc1 = {};
  float den = 0.0f;
  for (int q = 0; q < 13; q++) {
    v16h A;
    eval_pairgroup(4 * q + hA, [&](int e, float lw) {
      float wv = __expf(lw - mmax);
      den += wv;
      A[e] = (_Float16)wv;
    });
    eval_pairgroup(4 * q + 2 + hA, [&](int e, float lw) {
      float wv = __expf(lw - mmax);
      den += wv;
      A[8 + e] = (_Float16)wv;
    });

    u32 dv[16];
#pragma unroll
    for (int a = 0; a < 2; a++) {
      int p = 4 * q + bb + 2 * a;
      u32 ps = (u32)((p < 49) ? p : 0);
      int dx = (int)(ps / 7u) - 3;
      int dy = (int)(ps % 7u) - 3;
      int rowBase = ((clampi(ui + dx, 0, HL - 1) - h0) * 7 +
                     (clampi(vi + dy, 0, HL - 1) - w0)) * 8;
#pragma unroll
      for (int e = 0; e < 8; e++) {
        int bk = clampi(2 * wp + e - 3, 0, HL - 1) - z0;
        dv[a * 8 + e] = *(const u32*)(smem + waveBase + FEAT_OFF +
                                      ((rowBase + bk) << 6) + (n << 2));
      }
    }
    v16h B0, B1;
#pragma unroll
    for (int t = 0; t < 16; t++) {
      B0[t] = __builtin_bit_cast(_Float16, (unsigned short)(dv[t] & 0xffffu));
      B1[t] = __builtin_bit_cast(_Float16, (unsigned short)(dv[t] >> 16));
    }
    acc0 = __builtin_amdgcn_wmma_f32_16x16x32_f16(
        false, A, false, B0, (short)0, acc0, false, false);
    acc1 = __builtin_amdgcn_wmma_f32_16x16x32_f16(
        false, A, false, B1, (short)0, acc1, false, false);
  }

  // Denominator: merge complementary K halves, publish per voxel.
  den += __shfl_xor(den, 16, 32);
  if (lane < 16) {
    ((float*)(smem + waveBase + DEN_OFF))[lane] = den;
  }
  __syncthreads();

  // C/D layout: VGPR r, lanes 0-15 -> M=r, lanes 16-31 -> M=8+r; N = lane%16.
  const float* denArr = (const float*)(smem + waveBase + DEN_OFF);
#pragma unroll
  for (int r = 0; r < 8; r++) {
    int mm = r + ((lane < 16) ? 0 : 8);
    float inv = __builtin_amdgcn_rcpf(fmaxf(denArr[mm], 1e-8f));
    int ox2 = mm & 1, oy2 = (mm >> 1) & 1, oz4 = mm >> 2;
    int Xo = 2 * ui + ox2, Yo = 2 * vi + oy2, Zo = 4 * wp + oz4;
    size_t o = (size_t)Xo * 4096 + (size_t)Yo * 64 + (size_t)Zo;
    out[(size_t)n * 262144 + o] = acc0[r] * inv;
    out[(size_t)(n + 16) * 262144 + o] = acc1[r] * inv;
  }
}

// ---------------------------------------------------------------------------
extern "C" void kernel_launch(void* const* d_in, const int* in_sizes, int n_in,
                              void* d_out, int out_size, void* d_ws,
                              size_t ws_size, hipStream_t stream) {
  const float* feat = (const float*)d_in[0];   // (1,32,32,32,32)
  const float* guide = (const float*)d_in[1];  // (1,3,64,64,64)
  const float* sxr = (const float*)d_in[2];
  const float* syr = (const float*)d_in[3];
  const float* szr = (const float*)d_in[4];
  const float* srr = (const float*)d_in[5];
  float* out = (float*)d_out;                  // (1,32,64,64,64)

  unsigned char* ws = (unsigned char*)d_ws;    // needs >= 4 MiB
  float* sig = (float*)ws;                          // 1 MiB
  u32* featT = (u32*)(ws + (1u << 20));             // 2 MiB
  float* r2map = (float*)(ws + 3u * (1u << 20));    // 1 MiB

  prep_sig_kernel<<<128, 256, 0, stream>>>(sxr, syr, szr, srr, guide, sig);
  prep_r2_kernel<<<1024, 256, 0, stream>>>(sig, r2map);
  prep_feat_kernel<<<128, 256, 0, stream>>>(feat, featT);
  // 16384 tiles, 4 waves (tiles) per 128-thread block.
  jbu_main_kernel<<<4096, 128, DYN_LDS, stream>>>(guide, sig, featT, r2map, out);
}
